// EdgeToNodeAttention_87591563034737
// MI455X (gfx1250) — matrix operand order repaired and checked by reading, not compiled, not run on hardware
//
#include <hip/hip_runtime.h>
#include <math.h>

#define NNODES 50000
#define NEDGES 800000
#define DIM    128
#define NHEADS 8
#define CUTOFF 5.0f

typedef __attribute__((ext_vector_type(16))) __bf16 v16bf;
typedef __attribute__((ext_vector_type(8)))  float  v8f;

// ---------- monotone float <-> uint mapping for atomic segment-max ----------
__device__ __forceinline__ unsigned f2ord(float f) {
  unsigned u = __float_as_uint(f);
  return (u & 0x80000000u) ? ~u : (u | 0x80000000u);
}
__device__ __forceinline__ float ord2f(unsigned o) {
  return (o & 0x80000000u) ? __uint_as_float(o & 0x7fffffffu) : __uint_as_float(~o);
}

// ---------- bf16 fragment loaders (wave32 16x16x32 layout) ----------
// A (16x32, MxK): lanes 0-15 row M=lane, K = {0..15}; lanes 16-31 row
// M=lane-16, K = {16..31}. Split fp32 -> bf16 hi + bf16 lo residual.
__device__ __forceinline__ void load_afrag(const float* __restrict__ A, int row, int k0,
                                           int lane, v16bf* ahi, v16bf* alo) {
  const int half8 = (lane >> 4) << 4;                 // 0 or 16
  const float* p0 = A + (size_t)row * DIM + k0 + half8;
#pragma unroll
  for (int i = 0; i < 8; ++i) {
    float f0 = p0[i];
    float f1 = p0[8 + i];
    __bf16 h0 = (__bf16)f0;
    __bf16 h1 = (__bf16)f1;
    (*ahi)[i]     = h0;
    (*ahi)[8 + i] = h1;
    (*alo)[i]     = (__bf16)(f0 - (float)h0);
    (*alo)[8 + i] = (__bf16)(f1 - (float)h1);
  }
}

// B (32x16, KxN) from transposed weights in LDS: lane n = lane&15, K-half by
// lane>=16 -> 16 contiguous bf16 (32B) per lane.
__device__ __forceinline__ v16bf load_bfrag(const __bf16* __restrict__ sW, int n0, int k0, int lane) {
  const int n  = lane & 15;
  const int kh = (lane >> 4) << 4;                    // 0 or 16
  const __bf16* p = sW + (n0 + n) * DIM + k0 + kh;
  v16bf b;
#pragma unroll
  for (int i = 0; i < 16; ++i) b[i] = p[i];
  return b;
}

// Stage transposed hi/lo weight matrices (32KB each) into LDS.
__device__ __forceinline__ void stage_w(const __bf16* gHi, const __bf16* gLo,
                                        __bf16* sHi, __bf16* sLo) {
  const unsigned* srcH = (const unsigned*)gHi;
  const unsigned* srcL = (const unsigned*)gLo;
  unsigned* dstH = (unsigned*)sHi;
  unsigned* dstL = (unsigned*)sLo;
  for (int i = threadIdx.x; i < DIM * DIM / 2; i += (int)blockDim.x) {
    dstH[i] = srcH[i];
    dstL[i] = srcL[i];
  }
  __syncthreads();
}

// 16 rows x 128 cols GEMM strip: C = A[16x128] * W[128x128], bf16x3 split,
// f32 accumulate. 4 k-steps * 8 n-tiles * 3 = 96 WMMAs per wave.
__device__ __forceinline__ void gemm16x128(const float* __restrict__ A, int row0, int rowMax,
                                           const __bf16* sHi, const __bf16* sLo,
                                           int lane, v8f acc[8]) {
#pragma unroll
  for (int nt = 0; nt < 8; ++nt) { v8f z = {}; acc[nt] = z; }
  int r = row0 + (lane & 15);
  if (r > rowMax) r = rowMax;
#pragma unroll
  for (int kt = 0; kt < 4; ++kt) {
    const int k0 = kt * 32;
    v16bf ahi, alo;
    load_afrag(A, r, k0, lane, &ahi, &alo);
#pragma unroll
    for (int nt = 0; nt < 8; ++nt) {
      v16bf bhi = load_bfrag(sHi, nt * 16, k0, lane);
      v16bf blo = load_bfrag(sLo, nt * 16, k0, lane);
      acc[nt] = __builtin_amdgcn_wmma_f32_16x16x32_bf16(false, ahi, false, bhi, (short)0, acc[nt], false, false);
      acc[nt] = __builtin_amdgcn_wmma_f32_16x16x32_bf16(false, alo, false, bhi, (short)0, acc[nt], false, false);
      acc[nt] = __builtin_amdgcn_wmma_f32_16x16x32_bf16(false, ahi, false, blo, (short)0, acc[nt], false, false);
    }
  }
}

// ---------- prep: split all four weight matrices into transposed bf16 hi/lo ----------
__global__ void k_prep_weights(const float* __restrict__ WQ, const float* __restrict__ WK,
                               const float* __restrict__ WV, const float* __restrict__ WO,
                               __bf16* __restrict__ wt) {
  int t = blockIdx.x * blockDim.x + threadIdx.x;      // 0 .. 4*16384-1
  if (t >= 4 * DIM * DIM) return;
  int m   = t >> 14;
  int idx = t & (DIM * DIM - 1);
  int n = idx >> 7, k = idx & 127;
  const float* W = (m == 0) ? WQ : (m == 1) ? WK : (m == 2) ? WV : WO;
  float w = W[k * DIM + n];                            // transpose: Wt[n][k] = W[k][n]
  __bf16 hi = (__bf16)w;
  __bf16 lo = (__bf16)(w - (float)hi);
  __bf16* base = wt + (size_t)m * 2 * DIM * DIM;
  base[idx] = hi;
  base[DIM * DIM + idx] = lo;
}

__global__ void k_init(unsigned* __restrict__ segmax, float* __restrict__ segsum,
                       float* __restrict__ nodeacc) {
  const int stride = (int)(gridDim.x * blockDim.x);
  for (int i = blockIdx.x * blockDim.x + threadIdx.x; i < NNODES * NHEADS; i += stride) {
    segmax[i] = 0u;                                    // ord-min (== -inf)
    segsum[i] = 0.0f;
  }
  for (int i = blockIdx.x * blockDim.x + threadIdx.x; i < NNODES * DIM; i += stride) {
    nodeacc[i] = 0.0f;
  }
}

// ---------- node GEMM: out = A @ W + b (used for Q projection and O projection) ----------
__global__ void __launch_bounds__(256) k_node_gemm(const float* __restrict__ A, int nRows,
                                                   const __bf16* __restrict__ gW,
                                                   const float* __restrict__ bias,
                                                   float* __restrict__ out) {
  __shared__ __bf16 sHi[DIM * DIM];
  __shared__ __bf16 sLo[DIM * DIM];
  stage_w(gW, gW + DIM * DIM, sHi, sLo);
  const int lane = threadIdx.x & 31;
  const int wave = threadIdx.x >> 5;
  const int row0 = (blockIdx.x * 8 + wave) * 16;
  if (row0 >= nRows) return;                           // whole-wave exit: EXEC stays all-1 for WMMA
  v8f acc[8];
  gemm16x128(A, row0, nRows - 1, sHi, sLo, lane, acc);
  const int half = lane >> 4;
  const int n    = lane & 15;
#pragma unroll
  for (int h = 0; h < NHEADS; ++h) {
    const float bb = bias[h * 16 + n];
#pragma unroll
    for (int r = 0; r < 8; ++r) {
      const int row = row0 + r + (half << 3);
      if (row < nRows) out[(size_t)row * DIM + h * 16 + n] = acc[h][r] + bb;
    }
  }
}

// ---------- pass B: K = edge_attr@W_K+b_K (in regs), scores, atomic segment-max ----------
__global__ void __launch_bounds__(256) k_scores(const float* __restrict__ edge_attr,
                                                const int* __restrict__ edge_index,
                                                const float* __restrict__ Qn,
                                                const __bf16* __restrict__ gW,
                                                const float* __restrict__ bK,
                                                float* __restrict__ scores,
                                                unsigned* __restrict__ segmax) {
  __shared__ __bf16 sHi[DIM * DIM];
  __shared__ __bf16 sLo[DIM * DIM];
  stage_w(gW, gW + DIM * DIM, sHi, sLo);
  const int lane = threadIdx.x & 31;
  const int wave = threadIdx.x >> 5;
  const int e0   = (blockIdx.x * 8 + wave) * 16;
  if (e0 + 128 < NEDGES)                               // stream-ahead hint -> global_prefetch_b8
    __builtin_prefetch(edge_attr + (size_t)(e0 + 128 + (lane & 15)) * DIM, 0, 1);
  v8f acc[8];
  gemm16x128(edge_attr, e0, NEDGES - 1, sHi, sLo, lane, acc);
  const int half = lane >> 4;
  const int n    = lane & 15;
#pragma unroll
  for (int h = 0; h < NHEADS; ++h) {
    const float bb = bK[h * 16 + n];
#pragma unroll
    for (int r = 0; r < 8; ++r) acc[h][r] += bb;
  }
  const int* __restrict__ jrow = edge_index + NEDGES;  // edge_index[1][:]
#pragma unroll
  for (int r = 0; r < 8; ++r) {
    const int e = e0 + r + (half << 3);
    const int j = jrow[e];
    const float* qrow = Qn + (size_t)j * DIM;          // resident in 192MB L2
#pragma unroll
    for (int h = 0; h < NHEADS; ++h) {
      float v = acc[h][r] * qrow[h * 16 + n];          // d_k==16 -> head h == n-tile h
      v += __shfl_xor(v, 8, 32);                       // reduce within 16-lane half
      v += __shfl_xor(v, 4, 32);
      v += __shfl_xor(v, 2, 32);
      v += __shfl_xor(v, 1, 32);
      if (n == 0) {
        const float s = v * 0.25f;                     // 1/sqrt(16)
        scores[(size_t)e * NHEADS + h] = s;
        atomicMax(&segmax[(size_t)j * NHEADS + h], f2ord(s));
      }
    }
  }
}

// ---------- pass C: ex = exp(score - max), atomic segment-sum ----------
__global__ void k_expsum(float* __restrict__ scores, const int* __restrict__ edge_index,
                         const unsigned* __restrict__ segmax, float* __restrict__ segsum) {
  size_t t = (size_t)blockIdx.x * blockDim.x + threadIdx.x;
  if (t >= (size_t)NEDGES * NHEADS) return;
  size_t e = t >> 3;
  int h = (int)(t & 7);
  int j = edge_index[NEDGES + e];
  float m  = ord2f(segmax[(size_t)j * NHEADS + h]);
  float ex = expf(scores[t] - m);
  scores[t] = ex;                                      // overwrite in place
  atomicAdd(&segsum[(size_t)j * NHEADS + h], ex);
}

// ---------- pass D: V = edge_attr@W_V+b_V (in regs), scatter alpha*phi*V ----------
__global__ void __launch_bounds__(256) k_agg(const float* __restrict__ edge_attr,
                                             const int* __restrict__ edge_index,
                                             const float* __restrict__ edge_len,
                                             const __bf16* __restrict__ gW,
                                             const float* __restrict__ bV,
                                             const float* __restrict__ exv,
                                             const float* __restrict__ segsum,
                                             float* __restrict__ nodeacc) {
  __shared__ __bf16 sHi[DIM * DIM];
  __shared__ __bf16 sLo[DIM * DIM];
  stage_w(gW, gW + DIM * DIM, sHi, sLo);
  const int lane = threadIdx.x & 31;
  const int wave = threadIdx.x >> 5;
  const int e0   = (blockIdx.x * 8 + wave) * 16;
  if (e0 + 128 < NEDGES)
    __builtin_prefetch(edge_attr + (size_t)(e0 + 128 + (lane & 15)) * DIM, 0, 1);
  v8f acc[8];
  gemm16x128(edge_attr, e0, NEDGES - 1, sHi, sLo, lane, acc);
  const int half = lane >> 4;
  const int n    = lane & 15;
  const int* __restrict__ jrow = edge_index + NEDGES;
  const float kPi = 3.14159265358979f;
#pragma unroll
  for (int r = 0; r < 8; ++r) {
    const int e = e0 + r + (half << 3);
    const int j = jrow[e];
    const float phi = 0.5f * (cosf(kPi * edge_len[e] * (1.0f / CUTOFF)) + 1.0f);
#pragma unroll
    for (int h = 0; h < NHEADS; ++h) {
      const float w   = exv[(size_t)e * NHEADS + h] / segsum[(size_t)j * NHEADS + h] * phi;
      const float val = (acc[h][r] + bV[h * 16 + n]) * w;
      atomicAdd(&nodeacc[(size_t)j * DIM + h * 16 + n], val);  // L2-resident target
    }
  }
}

// ---------------------------------------------------------------------------
extern "C" void kernel_launch(void* const* d_in, const int* in_sizes, int n_in,
                              void* d_out, int out_size, void* d_ws, size_t ws_size,
                              hipStream_t stream) {
  const float* x          = (const float*)d_in[0];
  const int*   edge_index = (const int*)d_in[1];
  const float* edge_attr  = (const float*)d_in[2];
  const float* edge_len   = (const float*)d_in[3];
  const float* W_Q = (const float*)d_in[4];
  const float* b_Q = (const float*)d_in[5];
  const float* W_K = (const float*)d_in[6];
  const float* b_K = (const float*)d_in[7];
  const float* W_V = (const float*)d_in[8];
  const float* b_V = (const float*)d_in[9];
  const float* W_O = (const float*)d_in[10];
  const float* b_O = (const float*)d_in[11];

  // workspace layout (all 256B aligned), ~80.3 MB total
  char* ws = (char*)d_ws;
  float*    Qn      = (float*)(ws);                    // 50000*128*4 = 25,600,000
  float*    scores  = (float*)(ws + 25600000);         // 800000*8*4  = 25,600,000
  unsigned* segmax  = (unsigned*)(ws + 51200000);      // 50000*8*4   =  1,600,000
  float*    segsum  = (float*)(ws + 52800000);         // 50000*8*4   =  1,600,000
  float*    nodeacc = (float*)(ws + 54400000);         // 50000*128*4 = 25,600,000
  __bf16*   wt      = (__bf16*)(ws + 80000000);        // 4 * (hi+lo) * 16384 bf16 = 262,144

  const __bf16* wtQ = wt + 0 * 2 * DIM * DIM;
  const __bf16* wtK = wt + 1 * 2 * DIM * DIM;
  const __bf16* wtV = wt + 2 * 2 * DIM * DIM;
  const __bf16* wtO = wt + 3 * 2 * DIM * DIM;

  k_prep_weights<<<(4 * DIM * DIM + 255) / 256, 256, 0, stream>>>(W_Q, W_K, W_V, W_O, wt);
  k_init<<<512, 256, 0, stream>>>(segmax, segsum, nodeacc);

  const int nodeStrips = (NNODES + 15) / 16;           // 3125
  const int nodeBlocks = (nodeStrips + 7) / 8;         // 391
  const int edgeBlocks = NEDGES / 128;                 // 6250

  // Q projection on nodes
  k_node_gemm<<<nodeBlocks, 256, 0, stream>>>(x, NNODES, wtQ, b_Q, Qn);
  // fused K-GEMM + per-head scores + segment max
  k_scores<<<edgeBlocks, 256, 0, stream>>>(edge_attr, edge_index, Qn, wtK, b_K, scores, segmax);
  // softmax normalization pass
  k_expsum<<<(int)(((size_t)NEDGES * NHEADS + 255) / 256), 256, 0, stream>>>(scores, edge_index, segmax, segsum);
  // fused V-GEMM + alpha*phi weighting + scatter-add to nodes
  k_agg<<<edgeBlocks, 256, 0, stream>>>(edge_attr, edge_index, edge_len, wtV, b_V, scores, segsum, nodeacc);
  // output projection
  k_node_gemm<<<nodeBlocks, 256, 0, stream>>>(nodeacc, NNODES, wtO, b_O, (float*)d_out);
}
